// Graph_Conv_16449724743711
// MI455X (gfx1250) — matrix operand order repaired and checked by reading, not compiled, not run on hardware
//
#include <hip/hip_runtime.h>
#include <hip/hip_bf16.h>

// Problem constants (from reference)
#define BDIM 16
#define CDIM 64
#define VDIM 8192
#define KCH  5
#define ODIM 128
#define NJ   (BDIM * CDIM)   // 1024 = rows of Y = Xᵀ

typedef __attribute__((ext_vector_type(16))) __bf16 bf16x16;
typedef __attribute__((ext_vector_type(8)))  float  f32x8;

// CDNA5 async global->LDS path (ASYNCcnt-tracked DMA), with safe fallback.
#if defined(__AMDGCN__) && __has_builtin(__builtin_amdgcn_global_load_async_to_lds_b128) && \
    __has_builtin(__builtin_amdgcn_s_wait_asynccnt)
#define ASYNC_LDS 1
typedef int v4i_vs __attribute__((vector_size(16)));
using gv4i = __attribute__((address_space(1))) v4i_vs;   // global int4
using lv4i = __attribute__((address_space(3))) v4i_vs;   // LDS int4
#endif

static __device__ __forceinline__ void async_copy16(const unsigned short* g,
                                                    unsigned short* l) {
#ifdef ASYNC_LDS
    __builtin_amdgcn_global_load_async_to_lds_b128((gv4i*)g, (lv4i*)l, 0, 0);
#else
    *(uint4*)l = *(const uint4*)g;
#endif
}

static __device__ __forceinline__ void wait_async_all() {
#ifdef ASYNC_LDS
    __builtin_amdgcn_s_wait_asynccnt(0);
#endif
}

static __device__ __forceinline__ unsigned short f32_to_bf16_rne(float f) {
    union { float f; unsigned u; } v; v.f = f;
    unsigned u = v.u;
    if ((u & 0x7F800000u) == 0x7F800000u) {
        unsigned short h = (unsigned short)(u >> 16);
        if (u & 0x007FFFFFu) h |= 0x0040;   // quiet NaN
        return h;
    }
    u += 0x7FFFu + ((u >> 16) & 1u);        // round to nearest even
    return (unsigned short)(u >> 16);
}

// ---------------------------------------------------------------------------
// f32 -> bf16 bulk conversion (vectorized x4, grid-stride)
// ---------------------------------------------------------------------------
__global__ __launch_bounds__(256) void cvt_f32_bf16(const float* __restrict__ src,
                                                    unsigned short* __restrict__ dst,
                                                    long nvec /* n/4 */) {
    long i = (long)blockIdx.x * blockDim.x + threadIdx.x;
    long stride = (long)gridDim.x * blockDim.x;
    for (long p = i; p < nvec; p += stride) {
        float4 f = ((const float4*)src)[p];
        ushort4 o;
        o.x = f32_to_bf16_rne(f.x);
        o.y = f32_to_bf16_rne(f.y);
        o.z = f32_to_bf16_rne(f.z);
        o.w = f32_to_bf16_rne(f.w);
        ((ushort4*)dst)[p] = o;
    }
}

// ---------------------------------------------------------------------------
// Repack weight (OUT, C*K) -> Wt[k][o][c] so the update kernel reads rows.
// ---------------------------------------------------------------------------
__global__ void prep_w(const float* __restrict__ w, float* __restrict__ wt) {
    int i = blockIdx.x * blockDim.x + threadIdx.x;
    if (i >= KCH * ODIM * CDIM) return;
    int c = i % CDIM;
    int o = (i / CDIM) % ODIM;
    int k = i / (CDIM * ODIM);
    wt[i] = w[o * (CDIM * KCH) + c * KCH + k];
}

// ---------------------------------------------------------------------------
// Chebyshev GEMM:  Acc = Yprev(bf16, NJ x V) @ L(bf16, V x V, symmetric)
//   recur==0:  Ynew = Acc
//   recur==1:  Ynew = 2*Acc - prev2   (prev2 may alias outF: per-element RMW)
// Block tile 128x128, BK=64, 8 wave32 waves; wave tile 32x64 = 2x4 WMMA accs.
// Double-buffered LDS stages filled by GLOBAL_LOAD_ASYNC_TO_LDS_B128.
// ---------------------------------------------------------------------------
#define BM 128
#define BN 128
#define BK 64
#define PA 72   // padded LDS pitch (elems) -> conflict-free b128 frag reads
#define PB 72
#define STAGE_ELEMS (BM * PA + BN * PB)   // 18432 ushorts = 36,864 B per stage

__global__ __launch_bounds__(256) void cheb_gemm(
    const unsigned short* __restrict__ A,    // Yprev bf16  (NJ x VDIM)
    const unsigned short* __restrict__ Lb,   // L bf16      (VDIM x VDIM)
    const float* __restrict__ prev2,         // Y_{k-2} f32 (NJ x VDIM) or null
    float* __restrict__ outF,                // Y_k f32
    unsigned short* __restrict__ outB,       // Y_k bf16
    int recur)
{
    __shared__ unsigned short smem[2 * STAGE_ELEMS];   // 73,728 B (< 320 KB WGP LDS)

    const int tid  = threadIdx.x;
    const int n0   = blockIdx.x * BN;          // v block
    const int m0   = blockIdx.y * BM;          // j block
    const int lane = tid & 31;
    const int wid  = tid >> 5;
    const int wm   = (wid >> 1) * 32;          // wave row offset (4 waves in M)
    const int wn   = (wid & 1) * 64;           // wave col offset (2 waves in N)
    const int half = lane >> 4;
    const int l15  = lane & 15;

    // Issue one stage (A tile 128x64 + B tile 128x64) global -> LDS.
    auto issueStage = [&](int k0, int s) {
        unsigned short* sA = smem + s * STAGE_ELEMS;
        unsigned short* sB = sA + BM * PA;
#pragma unroll
        for (int i = 0; i < 4; ++i) {
            int lin = tid + i * 256;           // 1024 uint4 per tile
            int row = lin >> 3;                // 0..127
            int cg  = lin & 7;                 // 8 uint4 per 64-elem row
            async_copy16(A  + (size_t)(m0 + row) * VDIM + k0 + cg * 8,
                         sA + row * PA + cg * 8);
            async_copy16(Lb + (size_t)(n0 + row) * VDIM + k0 + cg * 8,
                         sB + row * PB + cg * 8);
        }
    };

    union Frag { bf16x16 v; uint4 u[2]; };
    // A-matrix 16-bit 16x32 layout (ISA 7.12.2): lanes<16 hold K{0..7,16..23},
    // lanes>=16 hold K{8..15,24..31}, M = lane&15.
    auto loadFragA = [&](const unsigned short* sA, int tm, int kk) -> bf16x16 {
        Frag r;
        const unsigned short* p = sA + (wm + tm * 16 + l15) * PA + kk + half * 8;
        r.u[0] = *(const uint4*)p;
        r.u[1] = *(const uint4*)(p + 16);
        return r.v;
    };
    // B-matrix 32x16: lanes<16 hold K{0..7,16..23}, lanes>=16 hold K{8..15,24..31}, N = lane&15.
    // sB holds L rows (= L columns, symmetric) so K is contiguous per N.
    auto loadFragB = [&](const unsigned short* sB, int tn, int kk) -> bf16x16 {
        Frag r;
        const unsigned short* p = sB + (wn + tn * 16 + l15) * PB + kk + half * 8;
        r.u[0] = *(const uint4*)p;
        r.u[1] = *(const uint4*)(p + 16);
        return r.v;
    };

    f32x8 acc[2][4];
#pragma unroll
    for (int tm = 0; tm < 2; ++tm)
#pragma unroll
        for (int tn = 0; tn < 4; ++tn)
            acc[tm][tn] = (f32x8){0.f, 0.f, 0.f, 0.f, 0.f, 0.f, 0.f, 0.f};

    const int steps = VDIM / BK;               // 128
    issueStage(0, 0);
    wait_async_all();
    __syncthreads();

    for (int kb = 0; kb < steps; ++kb) {
        const int cur = kb & 1;
        if (kb + 1 < steps) {
            issueStage((kb + 1) * BK, cur ^ 1);    // DMA overlaps the WMMAs below
            if (kb + 2 < steps) {                  // L2 prefetch two tiles ahead
                int kp = (kb + 2) * BK;
                __builtin_prefetch(A  + (size_t)(m0 + (tid >> 3)) * VDIM + kp, 0, 1);
                __builtin_prefetch(Lb + (size_t)(n0 + (tid >> 3)) * VDIM + kp, 0, 1);
            }
        }
        const unsigned short* sA = smem + cur * STAGE_ELEMS;
        const unsigned short* sB = sA + BM * PA;
#pragma unroll
        for (int kk = 0; kk < BK; kk += 32) {
            bf16x16 af[2], bfr[4];
            af[0]  = loadFragA(sA, 0, kk);
            af[1]  = loadFragA(sA, 1, kk);
            bfr[0] = loadFragB(sB, 0, kk);
            bfr[1] = loadFragB(sB, 1, kk);
            bfr[2] = loadFragB(sB, 2, kk);
            bfr[3] = loadFragB(sB, 3, kk);
#pragma unroll
            for (int tm = 0; tm < 2; ++tm)
#pragma unroll
                for (int tn = 0; tn < 4; ++tn)
                    acc[tm][tn] = __builtin_amdgcn_wmma_f32_16x16x32_bf16(
                        false, af[tm], false, bfr[tn],
                        (short)0, acc[tm][tn], false, false);
        }
        wait_async_all();     // next stage's DMA done (ran during the WMMAs)
        __syncthreads();      // all waves finished reading current stage
    }

    // Epilogue: C/D layout -> M = r + 8*half, N = lane&15 (per 16x16 tile)
#pragma unroll
    for (int tm = 0; tm < 2; ++tm) {
#pragma unroll
        for (int tn = 0; tn < 4; ++tn) {
            const int gm = m0 + wm + tm * 16 + half * 8;
            const int gn = n0 + wn + tn * 16 + l15;
#pragma unroll
            for (int r = 0; r < 8; ++r) {
                size_t idx = (size_t)(gm + r) * VDIM + gn;
                float a = acc[tm][tn][r];
                float val = recur ? (2.0f * a - prev2[idx]) : a;
                outF[idx] = val;
                outB[idx] = f32_to_bf16_rne(val);
            }
        }
    }
}

// ---------------------------------------------------------------------------
// Per-k projection: out[b,o,v] (+)= sum_c Y[b*64+c, v] * Wt[o,c]  (+bias @k=0)
// Block: one b, 32 consecutive v (lane = v -> coalesced), 8 waves x 16 o.
// ---------------------------------------------------------------------------
__global__ __launch_bounds__(256) void cheb_update(
    const float* __restrict__ Y,      // NJ x VDIM f32 (Y_k)
    const float* __restrict__ Wtk,    // ODIM x CDIM slice for this k
    const float* __restrict__ bias,
    float* __restrict__ out,          // B x ODIM x VDIM
    int addmode)
{
    __shared__ float xs[CDIM][33];           // 64 x 32 (+pad) X slice
    __shared__ float wsh[ODIM * CDIM];       // 128 x 64 weights

    const int tid = threadIdx.x;
    const int b   = blockIdx.y;
    const int v0  = blockIdx.x * 32;

    for (int i = tid; i < CDIM * 32; i += 256) {
        int c = i >> 5, v = i & 31;
        xs[c][v] = Y[(size_t)(b * CDIM + c) * VDIM + v0 + v];
    }
    for (int i = tid; i < ODIM * CDIM; i += 256) wsh[i] = Wtk[i];
    __syncthreads();

    const int v  = tid & 31;                 // lane -> v (coalesced I/O)
    const int ob = (tid >> 5) * 16;          // wave-uniform o range

    float acc[16];
    if (addmode) {
#pragma unroll
        for (int i = 0; i < 16; ++i)
            acc[i] = out[((size_t)b * ODIM + ob + i) * VDIM + v0 + v];
    } else {
#pragma unroll
        for (int i = 0; i < 16; ++i) acc[i] = bias[ob + i];
    }
    for (int c = 0; c < CDIM; ++c) {
        float xv = xs[c][v];
#pragma unroll
        for (int i = 0; i < 16; ++i)
            acc[i] = fmaf(xv, wsh[(ob + i) * CDIM + c], acc[i]);
    }
#pragma unroll
    for (int i = 0; i < 16; ++i)
        out[((size_t)b * ODIM + ob + i) * VDIM + v0 + v] = acc[i];
}

// ---------------------------------------------------------------------------
// Orchestration
// ---------------------------------------------------------------------------
extern "C" void kernel_launch(void* const* d_in, const int* in_sizes, int n_in,
                              void* d_out, int out_size, void* d_ws, size_t ws_size,
                              hipStream_t stream) {
    const float* x    = (const float*)d_in[0];   // (B,C,V) == Y0 (NJ x V) f32!
    const float* L    = (const float*)d_in[1];   // (V,V) symmetric
    const float* w    = (const float*)d_in[2];   // (OUT, C*K)
    const float* bias = (const float*)d_in[3];   // (OUT,)
    float* out = (float*)d_out;                  // (B, OUT, V)

    char* ws = (char*)d_ws;
    size_t off = 0;
    auto walloc = [&](size_t bytes) -> void* {
        void* p = ws + off;
        off = (off + bytes + 255) & ~(size_t)255;
        return p;
    };
    unsigned short* Lb  = (unsigned short*)walloc((size_t)VDIM * VDIM * 2); // 128 MB (L2-resident)
    unsigned short* Yb0 = (unsigned short*)walloc((size_t)NJ * VDIM * 2);
    unsigned short* Yb1 = (unsigned short*)walloc((size_t)NJ * VDIM * 2);
    float*          Yf0 = (float*)walloc((size_t)NJ * VDIM * 4);
    float*          Yf1 = (float*)walloc((size_t)NJ * VDIM * 4);
    float*          Wt  = (float*)walloc((size_t)KCH * ODIM * CDIM * 4);
    (void)ws_size; (void)in_sizes; (void)n_in; (void)out_size;

    const int OC = ODIM * CDIM;
    dim3 gg(VDIM / BN, NJ / BM);      // (64, 8) GEMM grid
    dim3 ug(VDIM / 32, BDIM);         // update grid

    // Precision prep: L -> bf16 (fits L2), Y0 -> bf16, weight repack
    cvt_f32_bf16<<<4096, 256, 0, stream>>>(L, Lb, (long)VDIM * VDIM / 4);
    cvt_f32_bf16<<<1024, 256, 0, stream>>>(x, Yb0, (long)NJ * VDIM / 4);
    prep_w<<<(KCH * OC + 255) / 256, 256, 0, stream>>>(w, Wt);

    // k=0 term (Y0 = input), init output with bias
    cheb_update<<<ug, 256, 0, stream>>>(x, Wt + 0 * OC, bias, out, 0);

    // k=1: Y1 = Y0 @ L
    cheb_gemm<<<gg, 256, 0, stream>>>(Yb0, Lb, nullptr, Yf0, Yb1, 0);
    cheb_update<<<ug, 256, 0, stream>>>(Yf0, Wt + 1 * OC, bias, out, 1);

    // k=2: Y2 = 2*(Y1 @ L) - Y0
    cheb_gemm<<<gg, 256, 0, stream>>>(Yb1, Lb, x, Yf1, Yb0, 1);
    cheb_update<<<ug, 256, 0, stream>>>(Yf1, Wt + 2 * OC, bias, out, 1);

    // k=3: Y3 = 2*(Y2 @ L) - Y1   (prev2 aliases dest: per-element RMW, safe)
    cheb_gemm<<<gg, 256, 0, stream>>>(Yb0, Lb, Yf0, Yf0, Yb1, 1);
    cheb_update<<<ug, 256, 0, stream>>>(Yf0, Wt + 3 * OC, bias, out, 1);

    // k=4: Y4 = 2*(Y3 @ L) - Y2
    cheb_gemm<<<gg, 256, 0, stream>>>(Yb1, Lb, Yf1, Yf1, Yb0, 1);
    cheb_update<<<ug, 256, 0, stream>>>(Yf1, Wt + 4 * OC, bias, out, 1);
}
